// QuantumNet_41283225649209
// MI455X (gfx1250) — matrix operand order repaired and checked by reading, not compiled, not run on hardware
//
#include <hip/hip_runtime.h>
#include <hip/hip_bf16.h>
#include <math.h>

typedef __attribute__((ext_vector_type(16))) _Float16 v16h;
typedef __attribute__((ext_vector_type(8)))  float    v8f;

static constexpr int NQUBITS = 10;
static constexpr int NLAYERS = 4;
static constexpr int NSTATE  = 1 << NQUBITS;   // 1024 amplitudes
static constexpr int HPAD    = 16;             // padded column stride for h

__device__ __forceinline__ float4 ld4(const float* p) {
  return *reinterpret_cast<const float4*>(p);
}
__device__ __forceinline__ void cvt4(v16h& d, int base, float4 v, float m) {
  d[base + 0] = (_Float16)(v.x * m);
  d[base + 1] = (_Float16)(v.y * m);
  d[base + 2] = (_Float16)(v.z * m);
  d[base + 3] = (_Float16)(v.w * m);
}

// ---------------------------------------------------------------------------
// Kernel 1: h = tanh(x @ Wp^T + bp) via v_wmma_f32_16x16x32_f16.
// One wave computes one 16(M) x 16(N) tile; N padded 10->16 via a multiply
// mask (no predicated loads), K split into 24 full chunks + 1 half tail so
// the steady-state loop has zero bounds checks and pure b128 loads.
// ---------------------------------------------------------------------------
__global__ __launch_bounds__(256) void proj_tanh_wmma(
    const float* __restrict__ x,    // (B, D)
    const float* __restrict__ Wp,   // (nq, D)
    const float* __restrict__ bp,   // (nq,)
    float* __restrict__ h,          // (B, HPAD)
    int B, int D, int nq)
{
  const int lane = threadIdx.x & 31;
  const int wave = threadIdx.x >> 5;
  const int m0   = (blockIdx.x * 8 + wave) * 16;
  if (m0 >= B) return;                          // wave-uniform

  const int group = lane >> 4;                  // lane half-group (0/1)
  const int lrow  = lane & 15;                  // M for A-frag, N for B-frag

  const float* xrow = x + (size_t)(m0 + lrow) * D;
  const int    wr   = (lrow < nq) ? lrow : (nq - 1);   // clamp, mask later
  const float  bmsk = (lrow < nq) ? 1.0f : 0.0f;
  const float* wrow = Wp + (size_t)wr * D;

  v8f acc = {};
  const int KFULL = (D / 32) * 32;              // 768 for D=784

  for (int kbase = 0; kbase < KFULL; kbase += 32) {
    __builtin_prefetch(xrow + kbase + 64, 0, 1);   // global_prefetch_b8

    // A fragment (16x32 f16): lane half-group holds K runs
    //   {8g..8g+7} -> elements 0..7, {16+8g..16+8g+7} -> elements 8..15
    float4 a0 = ld4(xrow + kbase + 8 * group);
    float4 a1 = ld4(xrow + kbase + 8 * group + 4);
    float4 a2 = ld4(xrow + kbase + 16 + 8 * group);
    float4 a3 = ld4(xrow + kbase + 16 + 8 * group + 4);
    // B fragment (32x16 f16): lane n=lrow holds K runs {8g..8g+7}, {16+8g..16+8g+7}
    float4 b0 = ld4(wrow + kbase + 8 * group);
    float4 b1 = ld4(wrow + kbase + 8 * group + 4);
    float4 b2 = ld4(wrow + kbase + 16 + 8 * group);
    float4 b3 = ld4(wrow + kbase + 16 + 8 * group + 4);

    v16h a, bf;
    cvt4(a, 0, a0, 1.0f);  cvt4(a, 4, a1, 1.0f);
    cvt4(a, 8, a2, 1.0f);  cvt4(a, 12, a3, 1.0f);
    cvt4(bf, 0, b0, bmsk); cvt4(bf, 4, b1, bmsk);
    cvt4(bf, 8, b2, bmsk); cvt4(bf, 12, b3, bmsk);

    acc = __builtin_amdgcn_wmma_f32_16x16x32_f16(
        false, a, false, bf, (short)0, acc, false, false);
  }

  // Tail chunk: K = KFULL..KFULL+15 valid (16 of 32), rest zero.
  if (KFULL < D) {
    // A: elements 0..7 have K = 8g+e in 0..15 (valid both groups); 8..15 zero.
    float4 a0 = ld4(xrow + KFULL + 8 * group);
    float4 a1 = ld4(xrow + KFULL + 8 * group + 4);
    // B: elements 0..7 have K = 8g+e in 0..15 (valid both groups); 8..15 zero.
    float4 b0 = ld4(wrow + KFULL + 8 * group);
    float4 b1 = ld4(wrow + KFULL + 8 * group + 4);

    v16h a = {}, bf = {};
    cvt4(a, 0, a0, 1.0f);  cvt4(a, 4, a1, 1.0f);
    cvt4(bf, 0, b0, bmsk); cvt4(bf, 4, b1, bmsk);

    acc = __builtin_amdgcn_wmma_f32_16x16x32_f16(
        false, a, false, bf, (short)0, acc, false, false);
  }

  // Epilogue: C/D layout: VGPR r -> M = r + 8*group, N = lrow.
  const float bias = (lrow < nq) ? bp[lrow] : 0.0f;
#pragma unroll
  for (int r = 0; r < 8; ++r) {
    int m = m0 + r + 8 * group;
    h[(size_t)m * HPAD + lrow] = tanhf(acc[r] + bias);
  }
}

// ---------------------------------------------------------------------------
// Kernel 2: 10-qubit statevector simulation, one block per batch element.
// Statevector (real/imag) lives entirely in LDS (8 KB). 256 threads handle
// 2 amplitude-pairs each per 1-qubit gate; CNOTs are LDS swaps.
// Bit convention: qubit q <-> bit q (pure axis relabeling vs. reference,
// applied consistently to embedding, Rot, CNOT ring and Z-expectations).
// ---------------------------------------------------------------------------
__global__ __launch_bounds__(256) void qsim_kernel(
    const float* __restrict__ h,    // (B, HPAD), already tanh'ed angles
    const float* __restrict__ qw,   // (NLAYERS, NQUBITS, 3)
    const float* __restrict__ Wo,   // (10, NQUBITS)
    const float* __restrict__ bo,   // (10,)
    float* __restrict__ out,        // (B, 10)
    int B)
{
  __shared__ float sr[NSTATE];
  __shared__ float si[NSTATE];
  __shared__ float zred[8 * NQUBITS];   // per-wave partial Z sums
  __shared__ float zf[NQUBITS];

  const int b   = blockIdx.x;
  const int tid = threadIdx.x;

  // |00...0>
  for (int i = tid; i < NSTATE; i += 256) { sr[i] = 0.0f; si[i] = 0.0f; }
  if (tid == 0) sr[0] = 1.0f;
  __syncthreads();

  // --- RY angle embedding (real 2x2 rotations) ---
  for (int q = 0; q < NQUBITS; ++q) {
    const float th = 0.5f * h[(size_t)b * HPAD + q];
    const float c = __cosf(th), s = __sinf(th);
#pragma unroll
    for (int t = 0; t < 2; ++t) {
      int p  = tid + t * 256;                                   // pair id 0..511
      int i0 = ((p >> q) << (q + 1)) | (p & ((1 << q) - 1));
      int i1 = i0 | (1 << q);
      float a0r = sr[i0], a0i = si[i0];
      float a1r = sr[i1], a1i = si[i1];
      sr[i0] = c * a0r - s * a1r;  si[i0] = c * a0i - s * a1i;
      sr[i1] = s * a0r + c * a1r;  si[i1] = s * a0i + c * a1i;
    }
    __syncthreads();
  }

  // --- Strongly entangling layers ---
  for (int l = 0; l < NLAYERS; ++l) {
    for (int q = 0; q < NQUBITS; ++q) {
      const float* w = qw + (l * NQUBITS + q) * 3;
      const float phi = w[0], the = w[1], omg = w[2];
      const float c = __cosf(0.5f * the), s = __sinf(0.5f * the);
      const float apo = 0.5f * (phi + omg), amo = 0.5f * (phi - omg);
      const float cpo = __cosf(apo), spo = __sinf(apo);
      const float cmo = __cosf(amo), smo = __sinf(amo);
      // Rot = RZ(omega) RY(theta) RZ(phi):
      const float u00r =  c * cpo, u00i = -c * spo;
      const float u01r = -s * cmo, u01i = -s * smo;
      const float u10r =  s * cmo, u10i = -s * smo;
      const float u11r =  c * cpo, u11i =  c * spo;
#pragma unroll
      for (int t = 0; t < 2; ++t) {
        int p  = tid + t * 256;
        int i0 = ((p >> q) << (q + 1)) | (p & ((1 << q) - 1));
        int i1 = i0 | (1 << q);
        float a0r = sr[i0], a0i = si[i0];
        float a1r = sr[i1], a1i = si[i1];
        sr[i0] = u00r * a0r - u00i * a0i + u01r * a1r - u01i * a1i;
        si[i0] = u00r * a0i + u00i * a0r + u01r * a1i + u01i * a1r;
        sr[i1] = u10r * a0r - u10i * a0i + u11r * a1r - u11i * a1i;
        si[i1] = u10r * a0i + u10i * a0r + u11r * a1i + u11i * a1r;
      }
      __syncthreads();
    }
    // CNOT ring: control q, target (q+1)%NQUBITS.
    for (int q = 0; q < NQUBITS; ++q) {
      const int cb = q, tb = (q + 1) % NQUBITS;
      const int lo = cb < tb ? cb : tb;
      const int hi = cb < tb ? tb : cb;
      int p = tid;                                              // 0..255
      int i = ((p >> lo) << (lo + 1)) | (p & ((1 << lo) - 1));
      i     = ((i >> hi) << (hi + 1)) | (i & ((1 << hi) - 1));
      i    |= (1 << cb);                                        // control=1, target=0
      int j = i | (1 << tb);
      float tr = sr[i], ti = si[i];
      sr[i] = sr[j]; si[i] = si[j];
      sr[j] = tr;    si[j] = ti;
      __syncthreads();
    }
  }

  // --- Z expectations: z[q] = sum_i |amp_i|^2 * (1 - 2*bit_q(i)) ---
  float part[NQUBITS];
#pragma unroll
  for (int q = 0; q < NQUBITS; ++q) part[q] = 0.0f;
  for (int t = 0; t < 4; ++t) {
    int i = tid + t * 256;
    float pr = sr[i] * sr[i] + si[i] * si[i];
#pragma unroll
    for (int q = 0; q < NQUBITS; ++q)
      part[q] += ((i >> q) & 1) ? -pr : pr;
  }
  // wave32 shuffle reduction, then 8 per-wave partials via LDS
#pragma unroll
  for (int q = 0; q < NQUBITS; ++q) {
    float v = part[q];
#pragma unroll
    for (int off = 16; off > 0; off >>= 1) v += __shfl_xor(v, off, 32);
    if ((tid & 31) == 0) zred[(tid >> 5) * NQUBITS + q] = v;
  }
  __syncthreads();
  if (tid < NQUBITS) {
    float z = 0.0f;
#pragma unroll
    for (int w = 0; w < 8; ++w) z += zred[w * NQUBITS + tid];
    zf[tid] = z;
  }
  __syncthreads();

  // --- Output projection: logits = z @ Wo^T + bo ---
  if (tid < 10) {
    float acc = bo[tid];
#pragma unroll
    for (int q = 0; q < NQUBITS; ++q) acc += zf[q] * Wo[tid * NQUBITS + q];
    out[(size_t)b * 10 + tid] = acc;
  }
}

// ---------------------------------------------------------------------------
extern "C" void kernel_launch(void* const* d_in, const int* in_sizes, int n_in,
                              void* d_out, int out_size, void* d_ws, size_t ws_size,
                              hipStream_t stream) {
  const float* x  = (const float*)d_in[0];  // (B, 784)
  const float* Wp = (const float*)d_in[1];  // (10, 784)
  const float* bp = (const float*)d_in[2];  // (10,)
  const float* qw = (const float*)d_in[3];  // (4, 10, 3)
  const float* Wo = (const float*)d_in[4];  // (10, 10)
  const float* bo = (const float*)d_in[5];  // (10,)
  float* out = (float*)d_out;               // (B, 10)

  const int D  = 784;
  const int nq = NQUBITS;
  const int Bn = in_sizes[0] / D;           // 8192

  float* hbuf = (float*)d_ws;               // (B, HPAD) fp32 = 512 KB

  const int tiles   = (Bn + 15) / 16;       // 512 16-row tiles
  const int blocks1 = (tiles + 7) / 8;      // 8 waves (tiles) per 256-thr block
  proj_tanh_wmma<<<blocks1, 256, 0, stream>>>(x, Wp, bp, hbuf, Bn, D, nq);

  qsim_kernel<<<Bn, 256, 0, stream>>>(hbuf, qw, Wo, bo, out, Bn);
}
